// SigmaCCS2_21019569947118
// MI455X (gfx1250) — matrix operand order repaired and checked by reading, not compiled, not run on hardware
//
#include <hip/hip_runtime.h>
#include <hip/hip_bf16.h>

typedef __attribute__((ext_vector_type(16))) _Float16 v16h;
typedef __attribute__((ext_vector_type(8)))  float    v8f;

#define WMMA_F16(a, b, c) \
  __builtin_amdgcn_wmma_f32_16x16x32_f16(false, (a), false, (b), (short)0, (c), false, false)

// ---------------------------------------------------------------------------
// Weight re-pack kernels: produce per-lane swizzled B fragments.
// Layout: elem index = ((c*4 + t)*32 + l)*16 + j ;  kk = 32c + (j&8)*2 + (l>>4)*8 + (j&7) ;
//         col = 16t + (l&15).   (B fragment layout per ISA 7.12.2, wave32)
// ---------------------------------------------------------------------------
__global__ void swz_w2_kernel(const float* __restrict__ w2, _Float16* __restrict__ out) {
  int idx = blockIdx.x * blockDim.x + threadIdx.x;          // 128*4*32*16 = 262144
  if (idx >= 128 * 4 * 32 * 16) return;
  int j = idx & 15, l = (idx >> 4) & 31, t = (idx >> 9) & 3, c = idx >> 11;
  int kk = 32 * c + ((j & 8) << 1) + ((l >> 4) << 3) + (j & 7); // 0..4095
  int k = kk >> 6, i = kk & 63;                             // kk = k*64 + i
  int o = (t << 4) + (l & 15);
  out[idx] = (_Float16)w2[(size_t)k * 4096 + i * 64 + o];   // B[kk,o] = w2[k, i*64+o]
}

__global__ void swz_m64_kernel(const float* __restrict__ mat, _Float16* __restrict__ out) {
  int idx = blockIdx.x * blockDim.x + threadIdx.x;          // 2*4*32*16 = 4096
  if (idx >= 2 * 4 * 32 * 16) return;
  int j = idx & 15, l = (idx >> 4) & 31, t = (idx >> 9) & 3, c = idx >> 11;
  int kk = 32 * c + ((j & 8) << 1) + ((l >> 4) << 3) + (j & 7); // 0..63
  int o = (t << 4) + (l & 15);
  out[idx] = (_Float16)mat[kk * 64 + o];
}

// ---------------------------------------------------------------------------
// Skinny input linears (K = 20/5 nodes, 4/1 edges): plain VALU, memory-bound.
// ---------------------------------------------------------------------------
__global__ void init_kernel(const float* __restrict__ in, const float* __restrict__ W,
                            const float* __restrict__ b, float* __restrict__ x, int N, int K) {
  int idx = blockIdx.x * blockDim.x + threadIdx.x;
  if (idx >= N * 64) return;
  int n = idx >> 6, o = idx & 63;
  float s = b[o];
  for (int k = 0; k < K; ++k) s += in[(size_t)n * K + k] * W[k * 64 + o];
  x[idx] = s > 0.f ? s : 0.f;
}

__global__ void edge_h_kernel(const float* __restrict__ ea, const float* __restrict__ W,
                              const float* __restrict__ b, _Float16* __restrict__ h, int E, int K) {
  int idx = blockIdx.x * blockDim.x + threadIdx.x;
  if (idx >= E * 64) return;
  int e = idx >> 6, o = idx & 63;
  float s = b[o];
  for (int k = 0; k < K; ++k) s += ea[(size_t)e * K + k] * W[k * 64 + o];
  h[idx] = (_Float16)(s > 0.f ? s : 0.f);
}

__global__ void zero_kernel(float* __restrict__ p, int n) {
  int i = blockIdx.x * blockDim.x + threadIdx.x;
  if (i < n) p[i] = 0.f;
}

__global__ void count_kernel(const int* __restrict__ tgt, float* __restrict__ cnt, int E) {
  int i = blockIdx.x * blockDim.x + threadIdx.x;
  if (i < E) atomicAdd(&cnt[tgt[i]], 1.f);
}

__global__ void inv_kernel(float* __restrict__ cnt, int N) {
  int i = blockIdx.x * blockDim.x + threadIdx.x;
  if (i < N) cnt[i] = 1.f / fmaxf(cnt[i], 1.f);
}

// ---------------------------------------------------------------------------
// msg_kernel: one wave per 32-edge tile (two 16-row A fragments sharing every
// B fragment -> 8 WMMAs per 8 b128 loads, and B is register double-buffered
// so the s_wait_loadcnt sits behind a full chunk of WMMAs + A-build VALU).
//   msg[e,:] = (h[e,:] (x) x[src[e],:]) @ W2'  +  x[src[e],:] @ B2
// ---------------------------------------------------------------------------
__global__ __launch_bounds__(32) void msg_kernel(
    const float* __restrict__ x, const _Float16* __restrict__ h,
    const _Float16* __restrict__ w2s, const _Float16* __restrict__ b2s,
    const int* __restrict__ src, const int* __restrict__ tgt,
    float* __restrict__ agg, int E) {
  __shared__ float    xs[32][64];
  __shared__ _Float16 hs[32][64];
  const int l = threadIdx.x;
  const int ebase = blockIdx.x * 32;

  { // cooperative gather: one row per lane (256B x, 128B h)
    int e = ebase + l;
    int s = src[e];
    const float4* xr = (const float4*)(x + (size_t)s * 64);
    float4* xd = (float4*)&xs[l][0];
#pragma unroll
    for (int q = 0; q < 16; ++q) xd[q] = xr[q];
    const uint4* hr = (const uint4*)(h + (size_t)e * 64);
    uint4* hd = (uint4*)&hs[l][0];
#pragma unroll
    for (int q = 0; q < 8; ++q) hd[q] = hr[q];
  }
  __syncthreads();

  const int m = l & 15;   // A-matrix row within a 16-edge fragment
  const int hi = l >> 4;  // lane half

  // Pre-converted f16 x fragments: xv[u][par], u = edge fragment, par = i-base/32.
  // A element j: j<8 -> K = hi*8+j ; j>=8 -> K = 16+hi*8+(j-8); i = par*32 + K.
  v16h xv[2][2];
#pragma unroll
  for (int u = 0; u < 2; ++u)
#pragma unroll
    for (int par = 0; par < 2; ++par)
#pragma unroll
      for (int j = 0; j < 8; ++j) {
        xv[u][par][j]     = (_Float16)xs[u * 16 + m][par * 32 + hi * 8 + j];
        xv[u][par][8 + j] = (_Float16)xs[u * 16 + m][par * 32 + 16 + hi * 8 + j];
      }

  v8f a00 = {}, a01 = {}, a02 = {}, a03 = {};  // edge fragment 0, N tiles 0..3
  v8f a10 = {}, a11 = {}, a12 = {}, a13 = {};  // edge fragment 1

  const v16h* bbase = (const v16h*)w2s + l;    // fragment idx = (c*4 + t)*32 + l
  v16h b0 = bbase[0], b1 = bbase[32], b2v = bbase[64], b3 = bbase[96];

#pragma unroll 1
  for (int k = 0; k < 64; ++k) {               // chunk pair: c = 2k (par0), 2k+1 (par1)
    const _Float16 h0 = hs[m][k];
    const _Float16 h1 = hs[16 + m][k];

    { // chunk c = 2k, parity 0
      v16h q0, q1;
#pragma unroll
      for (int j = 0; j < 16; ++j) { q0[j] = h0 * xv[0][0][j]; q1[j] = h1 * xv[1][0][j]; }
      const v16h* nb = bbase + (size_t)(2 * k + 1) * 128;  // prefetch next chunk
      v16h n0 = nb[0], n1 = nb[32], n2 = nb[64], n3 = nb[96];
      a00 = WMMA_F16(q0, b0, a00);  a10 = WMMA_F16(q1, b0, a10);
      a01 = WMMA_F16(q0, b1, a01);  a11 = WMMA_F16(q1, b1, a11);
      a02 = WMMA_F16(q0, b2v, a02); a12 = WMMA_F16(q1, b2v, a12);
      a03 = WMMA_F16(q0, b3, a03);  a13 = WMMA_F16(q1, b3, a13);
      b0 = n0; b1 = n1; b2v = n2; b3 = n3;
    }
    { // chunk c = 2k+1, parity 1
      v16h q0, q1;
#pragma unroll
      for (int j = 0; j < 16; ++j) { q0[j] = h0 * xv[0][1][j]; q1[j] = h1 * xv[1][1][j]; }
      const v16h* nb = bbase + (size_t)((2 * k + 2) & 127) * 128;  // wraps on last iter
      v16h n0 = nb[0], n1 = nb[32], n2 = nb[64], n3 = nb[96];
      a00 = WMMA_F16(q0, b0, a00);  a10 = WMMA_F16(q1, b0, a10);
      a01 = WMMA_F16(q0, b1, a01);  a11 = WMMA_F16(q1, b1, a11);
      a02 = WMMA_F16(q0, b2v, a02); a12 = WMMA_F16(q1, b2v, a12);
      a03 = WMMA_F16(q0, b3, a03);  a13 = WMMA_F16(q1, b3, a13);
      b0 = n0; b1 = n1; b2v = n2; b3 = n3;
    }
  }

  // + x_src @ B2 (b2 reshaped 64x64); A is xv directly (already f16)
#pragma unroll
  for (int c = 0; c < 2; ++c) {
    const v16h* bp = (const v16h*)b2s + (size_t)(c * 4) * 32 + l;
    v16h c0 = bp[0], c1 = bp[32], c2 = bp[64], c3 = bp[96];
    a00 = WMMA_F16(xv[0][c], c0, a00);  a10 = WMMA_F16(xv[1][c], c0, a10);
    a01 = WMMA_F16(xv[0][c], c1, a01);  a11 = WMMA_F16(xv[1][c], c1, a11);
    a02 = WMMA_F16(xv[0][c], c2, a02);  a12 = WMMA_F16(xv[1][c], c2, a12);
    a03 = WMMA_F16(xv[0][c], c3, a03);  a13 = WMMA_F16(xv[1][c], c3, a13);
  }

  // Scatter-add: lane l holds rows r + 8*hi (per fragment), col n = l&15 (+16/tile)
  const int n = l & 15;
#pragma unroll
  for (int r = 0; r < 8; ++r) {
    int t0 = tgt[ebase + r + 8 * hi];
    float* base0 = agg + (size_t)t0 * 64 + n;
    atomicAdd(base0 +  0, a00[r]);
    atomicAdd(base0 + 16, a01[r]);
    atomicAdd(base0 + 32, a02[r]);
    atomicAdd(base0 + 48, a03[r]);
    int t1 = tgt[ebase + 16 + r + 8 * hi];
    float* base1 = agg + (size_t)t1 * 64 + n;
    atomicAdd(base1 +  0, a10[r]);
    atomicAdd(base1 + 16, a11[r]);
    atomicAdd(base1 + 32, a12[r]);
    atomicAdd(base1 + 48, a13[r]);
  }
}

// ---------------------------------------------------------------------------
// update_kernel: x = relu(x @ root + agg*inv + bias), one wave per 16 nodes.
// ---------------------------------------------------------------------------
__global__ __launch_bounds__(32) void update_kernel(
    float* __restrict__ x, const float* __restrict__ agg, const float* __restrict__ inv,
    const float* __restrict__ bias, const _Float16* __restrict__ rootswz, int N) {
  __shared__ float xs[16][64];
  const int l = threadIdx.x;
  const int nb = blockIdx.x * 16;
  {
    int row = l >> 1, hsel = l & 1;
    const float4* xr = (const float4*)(x + (size_t)(nb + row) * 64 + hsel * 32);
    float4* xd = (float4*)&xs[row][hsel * 32];
#pragma unroll
    for (int q = 0; q < 8; ++q) xd[q] = xr[q];
  }
  __syncthreads();

  const int m = l & 15, hi = l >> 4;
  v16h a0, a1;
#pragma unroll
  for (int j = 0; j < 8; ++j) {
    a0[j]     = (_Float16)xs[m][hi * 8 + j];
    a0[8 + j] = (_Float16)xs[m][16 + hi * 8 + j];
    a1[j]     = (_Float16)xs[m][32 + hi * 8 + j];
    a1[8 + j] = (_Float16)xs[m][48 + hi * 8 + j];
  }
  v8f acc0 = {}, acc1 = {}, acc2 = {}, acc3 = {};
#pragma unroll
  for (int c = 0; c < 2; ++c) {
    v16h a = c ? a1 : a0;
    const v16h* bp = (const v16h*)rootswz + (size_t)(c * 4) * 32 + l;
    acc0 = WMMA_F16(a, bp[0],  acc0);
    acc1 = WMMA_F16(a, bp[32], acc1);
    acc2 = WMMA_F16(a, bp[64], acc2);
    acc3 = WMMA_F16(a, bp[96], acc3);
  }
  const int n = l & 15;
#pragma unroll
  for (int r = 0; r < 8; ++r) {
    int node = nb + r + 8 * hi;
    float iv = inv[node];
    float* xo = x + (size_t)node * 64;
    const float* ag = agg + (size_t)node * 64;
    float v0 = acc0[r] + ag[n]      * iv + bias[n];
    float v1 = acc1[r] + ag[n + 16] * iv + bias[n + 16];
    float v2 = acc2[r] + ag[n + 32] * iv + bias[n + 32];
    float v3 = acc3[r] + ag[n + 48] * iv + bias[n + 48];
    xo[n]      = v0 > 0.f ? v0 : 0.f;
    xo[n + 16] = v1 > 0.f ? v1 : 0.f;
    xo[n + 32] = v2 > 0.f ? v2 : 0.f;
    xo[n + 48] = v3 > 0.f ? v3 : 0.f;
  }
}

// ---------------------------------------------------------------------------
// Sum-pool over nodes -> 64 floats.
// ---------------------------------------------------------------------------
__global__ __launch_bounds__(256) void pool_kernel(const float* __restrict__ x,
                                                   float* __restrict__ out64, int N) {
  __shared__ float red[256];
  const int o = threadIdx.x & 63, rr = threadIdx.x >> 6;
  float s = 0.f;
  for (int n = blockIdx.x * 4 + rr; n < N; n += gridDim.x * 4) s += x[(size_t)n * 64 + o];
  red[threadIdx.x] = s;
  __syncthreads();
  if (rr == 0) atomicAdd(&out64[o], red[o] + red[64 + o] + red[128 + o] + red[192 + o]);
}

// ---------------------------------------------------------------------------
// Head MLP: [131] -> 384 -> (6x 384) -> 1. Single block, trivial FLOPs.
// ---------------------------------------------------------------------------
__global__ __launch_bounds__(384) void head_kernel(
    const float* __restrict__ pg, const float* __restrict__ plg, const float* __restrict__ adduct,
    const float* __restrict__ bw, const float* __restrict__ bb,
    const float* __restrict__ w1, const float* __restrict__ b1,
    const float* __restrict__ w2, const float* __restrict__ b2,
    float* __restrict__ out) {
  __shared__ float vin[131];
  __shared__ float va[384], vb[384], red[384];
  const int t = threadIdx.x;
  if (t < 64) vin[t] = pg[t];
  else if (t < 128) vin[t] = plg[t - 64];
  else if (t < 131) vin[t] = adduct[t - 128];
  __syncthreads();
  float s = bb[t];
  for (int k = 0; k < 131; ++k) s += vin[k] * bw[k * 384 + t];
  va[t] = fmaxf(s, 0.f);
  __syncthreads();
  float* cur = va; float* nxt = vb;
  for (int it = 0; it < 6; ++it) {
    float s2 = b1[t];
    for (int k = 0; k < 384; ++k) s2 += cur[k] * w1[k * 384 + t];
    nxt[t] = fmaxf(s2, 0.f);
    __syncthreads();
    float* tmp = cur; cur = nxt; nxt = tmp;
  }
  red[t] = cur[t] * w2[t];
  __syncthreads();
  for (int sN = 192; sN >= 3; sN >>= 1) {
    if (t < sN) red[t] += red[t + sN];
    __syncthreads();
  }
  if (t == 0) out[0] = red[0] + red[1] + red[2] + b2[0];
}

// ---------------------------------------------------------------------------
extern "C" void kernel_launch(void* const* d_in, const int* in_sizes, int n_in,
                              void* d_out, int out_size, void* d_ws, size_t ws_size,
                              hipStream_t stream) {
  const float* gx      = (const float*)d_in[0];
  const int*   g_ei    = (const int*)  d_in[1];
  const float* g_ea    = (const float*)d_in[2];
  const float* lgx     = (const float*)d_in[3];
  const int*   lg_ei   = (const int*)  d_in[4];
  const float* lg_ea   = (const float*)d_in[5];
  const float* adduct  = (const float*)d_in[6];
  const float* lin0_w  = (const float*)d_in[7];
  const float* lin0_b  = (const float*)d_in[8];
  const float* g_w1    = (const float*)d_in[9];
  const float* g_b1    = (const float*)d_in[10];
  const float* g_w2    = (const float*)d_in[11];
  const float* g_b2    = (const float*)d_in[12];
  const float* g_root  = (const float*)d_in[13];
  const float* g_bias  = (const float*)d_in[14];
  const float* lin0l_w = (const float*)d_in[15];
  const float* lin0l_b = (const float*)d_in[16];
  const float* lg_w1   = (const float*)d_in[17];
  const float* lg_b1   = (const float*)d_in[18];
  const float* lg_w2   = (const float*)d_in[19];
  const float* lg_b2   = (const float*)d_in[20];
  const float* lg_root = (const float*)d_in[21];
  const float* lg_bias = (const float*)d_in[22];
  const float* bott_w  = (const float*)d_in[23];
  const float* bott_b  = (const float*)d_in[24];
  const float* lin1_w  = (const float*)d_in[25];
  const float* lin1_b  = (const float*)d_in[26];
  const float* lin2_w  = (const float*)d_in[27];
  const float* lin2_b  = (const float*)d_in[28];

  const int NG = 30000, NLG = 60000, E = 60000;

  char* p = (char*)d_ws;
  auto alloc = [&](size_t bytes) -> void* {
    void* r = (void*)p;
    p += (bytes + 255) & ~(size_t)255;
    return r;
  };
  float*     x_g    = (float*)    alloc((size_t)NG  * 64 * 4);
  float*     x_lg   = (float*)    alloc((size_t)NLG * 64 * 4);
  float*     agg_g  = (float*)    alloc((size_t)NG  * 64 * 4);
  float*     agg_lg = (float*)    alloc((size_t)NLG * 64 * 4);
  _Float16*  h_g    = (_Float16*) alloc((size_t)E * 64 * 2);
  _Float16*  h_lg   = (_Float16*) alloc((size_t)E * 64 * 2);
  _Float16*  w2s_g  = (_Float16*) alloc(262144 * 2);
  _Float16*  w2s_lg = (_Float16*) alloc(262144 * 2);
  _Float16*  b2s_g  = (_Float16*) alloc(4096 * 2);
  _Float16*  b2s_lg = (_Float16*) alloc(4096 * 2);
  _Float16*  rts_g  = (_Float16*) alloc(4096 * 2);
  _Float16*  rts_lg = (_Float16*) alloc(4096 * 2);
  float*     inv_g  = (float*)    alloc((size_t)NG  * 4);
  float*     inv_lg = (float*)    alloc((size_t)NLG * 4);
  float*     pooled = (float*)    alloc(128 * 4);

  // weight re-pack (once per call; cheap)
  swz_w2_kernel <<<262144 / 256, 256, 0, stream>>>(g_w2,  w2s_g);
  swz_w2_kernel <<<262144 / 256, 256, 0, stream>>>(lg_w2, w2s_lg);
  swz_m64_kernel<<<16, 256, 0, stream>>>(g_b2,    b2s_g);
  swz_m64_kernel<<<16, 256, 0, stream>>>(lg_b2,   b2s_lg);
  swz_m64_kernel<<<16, 256, 0, stream>>>(g_root,  rts_g);
  swz_m64_kernel<<<16, 256, 0, stream>>>(lg_root, rts_lg);

  auto run_branch = [&](const float* X, int indim, int N, const int* ei, const float* ea, int eadim,
                        const float* l0w, const float* l0b, const float* w1, const float* b1,
                        _Float16* w2s, _Float16* b2s, _Float16* rts, const float* bias,
                        float* x, float* agg, _Float16* h, float* inv, float* pout) {
    const int* src = ei;
    const int* tgt = ei + E;
    init_kernel  <<<(N * 64 + 255) / 256, 256, 0, stream>>>(X, l0w, l0b, x, N, indim);
    edge_h_kernel<<<(E * 64 + 255) / 256, 256, 0, stream>>>(ea, w1, b1, h, E, eadim);
    zero_kernel  <<<(N + 255) / 256, 256, 0, stream>>>(inv, N);
    count_kernel <<<(E + 255) / 256, 256, 0, stream>>>(tgt, inv, E);
    inv_kernel   <<<(N + 255) / 256, 256, 0, stream>>>(inv, N);
    for (int it = 0; it < 3; ++it) {
      zero_kernel  <<<(N * 64 + 255) / 256, 256, 0, stream>>>(agg, N * 64);
      msg_kernel   <<<E / 32, 32, 0, stream>>>(x, h, w2s, b2s, src, tgt, agg, E);
      update_kernel<<<N / 16, 32, 0, stream>>>(x, agg, inv, bias, rts, N);
    }
    zero_kernel<<<1, 64, 0, stream>>>(pout, 64);
    pool_kernel<<<128, 256, 0, stream>>>(x, pout, N);
  };

  run_branch(gx, 20, NG, g_ei, g_ea, 4, lin0_w, lin0_b, g_w1, g_b1,
             w2s_g, b2s_g, rts_g, g_bias, x_g, agg_g, h_g, inv_g, pooled);
  run_branch(lgx, 5, NLG, lg_ei, lg_ea, 1, lin0l_w, lin0l_b, lg_w1, lg_b1,
             w2s_lg, b2s_lg, rts_lg, lg_bias, x_lg, agg_lg, h_lg, inv_lg, pooled + 64);

  head_kernel<<<1, 384, 0, stream>>>(pooled, pooled + 64, adduct,
                                     bott_w, bott_b, lin1_w, lin1_b, lin2_w, lin2_b,
                                     (float*)d_out);
}
